// ResidueEncoder_16758962389292
// MI455X (gfx1250) — compile-verified, matrix operand order checked by reading, not run on hardware
//
#include <hip/hip_runtime.h>
#include <hip/hip_bf16.h>

// ---------------- problem constants (match reference) ----------------
#define NNODES 30000
#define NEDGES 480000
#define F_INP  1280
#define HC     512           // HEADS * C  (GAT concat output width)
#define HEADS  4
#define HID    64
#define ETOT   (NEDGES + NNODES)   // edges + self loops

typedef __attribute__((ext_vector_type(8)))  __bf16 v8bf;
typedef __attribute__((ext_vector_type(16))) __bf16 v16bf;
typedef __attribute__((ext_vector_type(8)))  float  v8f;

// ---------------- helpers ----------------
__device__ __forceinline__ unsigned short f2bf(float f) {
  unsigned int u = __float_as_uint(f);
  u += 0x7fffu + ((u >> 16) & 1u);       // round-to-nearest-even
  return (unsigned short)(u >> 16);
}

__device__ __forceinline__ float atomic_max_f32(float* addr, float val) {
  // works with -inf init: positive path via signed max, negative via unsigned min
  if (val >= 0.0f)
    return __int_as_float(atomicMax((int*)addr, __float_as_int(val)));
  return __uint_as_float(atomicMin((unsigned int*)addr, __float_as_uint(val)));
}

__device__ __forceinline__ void edge_sd(const int* __restrict__ ei, int e, int& s, int& d) {
  if (e < NEDGES) { s = ei[e]; d = ei[NEDGES + e]; }
  else            { s = d = e - NEDGES; }              // self-loop
}

// load one 16x32-bf16 WMMA fragment half-row: 2x16B from (p) and (p+16 elems)
__device__ __forceinline__ v16bf make_frag(const unsigned short* p) {
  v8bf c0 = *(const v8bf*)p;
  v8bf c1 = *(const v8bf*)(p + 16);
  return __builtin_shufflevector(c0, c1, 0,1,2,3,4,5,6,7,8,9,10,11,12,13,14,15);
}

// CDNA5 async copy: global -> LDS, 16 bytes per active lane, tracked by ASYNCcnt
__device__ __forceinline__ void async_ld_b128(unsigned lds_off, const void* gaddr) {
  asm volatile("global_load_async_to_lds_b128 %0, %1, off"
               :: "v"(lds_off), "v"(gaddr) : "memory");
}
__device__ __forceinline__ void wait_async0() {
  asm volatile("s_wait_asynccnt 0x0" ::: "memory");
}

// ---------------- bf16 WMMA GEMM: C[MxN] = A[MxK] * B^T[NxK] ----------------
// Block = 128 threads = 4 waves. Each wave computes a 16x(WT*16) strip.
// STAGE: double-buffered async staging of the shared 16x32 A tile in LDS;
// loads for iteration i+1 overlap the WMMAs of iteration i.
// Requires M%16==0, N % (4*WT*16) == 0 per grid.x, K%32==0.
template<int WT, bool STAGE>
__global__ __launch_bounds__(128) void k_gemm_bf16_wmma(
    const unsigned short* __restrict__ A,   // M x K row-major bf16
    const unsigned short* __restrict__ B,   // N x K row-major bf16 (pre-transposed weight)
    float* __restrict__ C,                  // M x N row-major f32
    int M, int N, int K)
{
  __shared__ __align__(16) unsigned short atile[2][16 * 32];  // 2 x 1KB

  const int tid   = threadIdx.x;
  const int lane  = tid & 31;
  const int wave  = tid >> 5;
  const int mbase = blockIdx.y * 16;
  const int nbase = blockIdx.x * (4 * WT * 16) + wave * (WT * 16);
  const int lrow  = lane & 15;
  const int khalf = (lane >> 4) * 8;      // lanes 0-15: K {0-7,16-23}; 16-31: K {8-15,24-31}

  v8f acc[WT] = {};

  // direct-load path (used when !STAGE)
  const unsigned short* arow = A + (size_t)(mbase + lrow) * K + khalf;

  // staging path: 64 lanes (waves 0,1) each move one 16B chunk of the 16x32 tile
  //   chunk tid -> row tid>>2, k-sub (tid&3)*8 ; LDS byte offset tid*16
  const unsigned short* gsrc =
      A + (size_t)(mbase + (tid >> 2)) * K + (tid & 3) * 8;
  const unsigned ldsA0 = (unsigned)(size_t)(void*)&atile[0][0] + (unsigned)tid * 16u;
  const unsigned ldsA1 = (unsigned)(size_t)(void*)&atile[1][0] + (unsigned)tid * 16u;

  const int nIter = K / 32;
  if (STAGE && tid < 64) async_ld_b128(ldsA0, gsrc);          // prologue: k0=0 -> buf0

  for (int it = 0; it < nIter; ++it) {
    const int k0 = it * 32;
    v16bf afrag;
    if (STAGE) {
      wait_async0();                 // our buf[it&1] chunks have landed (waves 0,1)
      __syncthreads();               // everyone done reading buf[(it-1)&1]; data visible
      if (tid < 64 && it + 1 < nIter)
        async_ld_b128((it & 1) ? ldsA0 : ldsA1, gsrc + k0 + 32);  // overlap next tile
      afrag = make_frag(&atile[it & 1][lrow * 32 + khalf]);
    } else {
      afrag = make_frag(arow + k0);
      if (k0 + 32 < K) __builtin_prefetch(arow + k0 + 32, 0, 0); // global_prefetch_b8
    }
#pragma unroll
    for (int t = 0; t < WT; ++t) {
      const unsigned short* bp =
          B + (size_t)(nbase + t * 16 + lrow) * K + k0 + khalf;
      v16bf bfrag = make_frag(bp);
      acc[t] = __builtin_amdgcn_wmma_f32_16x16x32_bf16(
          false, afrag, false, bfrag, (short)0, acc[t], false, false);
    }
  }

  // C/D layout: VGPR r -> M = mbase + r + 8*(lane>=16), N = nbase + t*16 + (lane&15)
  const int crow0 = mbase + ((lane >> 4) * 8);
  const int ccol0 = nbase + (lane & 15);
#pragma unroll
  for (int t = 0; t < WT; ++t) {
#pragma unroll
    for (int r = 0; r < 8; ++r) {
      C[(size_t)(crow0 + r) * N + (ccol0 + t * 16)] = acc[t][r];
    }
  }
}

// ---------------- casts / transposes ----------------
__global__ void k_cast_bf16(const float* __restrict__ in, unsigned short* __restrict__ o, int n) {
  int i = blockIdx.x * blockDim.x + threadIdx.x;
  if (i < n) o[i] = f2bf(in[i]);
}

// out[n*K + k] = bf16(in[k*Nn + n])  (weight transpose for WMMA B operand)
__global__ void k_transpose_cast(const float* __restrict__ in, unsigned short* __restrict__ o,
                                 int Nn, int K) {
  int i = blockIdx.x * blockDim.x + threadIdx.x;
  if (i < Nn * K) {
    int n = i / K, k = i - n * K;
    o[i] = f2bf(in[(size_t)k * Nn + n]);
  }
}

// ---------------- attention coefficients: a = <h[n,h,:], att[h,:]> ----------------
__global__ __launch_bounds__(512) void k_att(const float* __restrict__ h,
    const float* __restrict__ wsrc, const float* __restrict__ wdst,
    float* __restrict__ a_src, float* __restrict__ a_dst)
{
  __shared__ float sA[16], sB[16];
  int n = blockIdx.x, t = threadIdx.x;
  float v = h[(size_t)n * HC + t];
  float p = v * wsrc[t], q = v * wdst[t];
  for (int m = 16; m > 0; m >>= 1) { p += __shfl_xor(p, m, 32); q += __shfl_xor(q, m, 32); }
  if ((t & 31) == 0) { sA[t >> 5] = p; sB[t >> 5] = q; }
  __syncthreads();
  if (t < HEADS) {
    float s = 0.f, d = 0.f;
    for (int j = 0; j < 4; ++j) { s += sA[t * 4 + j]; d += sB[t * 4 + j]; }
    a_src[n * HEADS + t] = s;
    a_dst[n * HEADS + t] = d;
  }
}

// ---------------- init kernels ----------------
__global__ void k_init_node(float* maxb, float* sumb, float* deg, float* bnsum, float* bnsq) {
  int i = blockIdx.x * blockDim.x + threadIdx.x;
  if (i < NNODES * HEADS) { maxb[i] = __uint_as_float(0xff800000u); sumb[i] = 0.f; }
  if (i < NNODES) deg[i] = 0.f;
  if (i < HC) { bnsum[i] = 0.f; bnsq[i] = 0.f; }
}

__global__ void k_init_out1(float* out1, const float* __restrict__ b) {
  int i = blockIdx.x * blockDim.x + threadIdx.x;
  if (i < NNODES * HC) out1[i] = b[i & (HC - 1)];
}

__global__ void k_init_out(float* out, const float* __restrict__ b) {
  int i = blockIdx.x * blockDim.x + threadIdx.x;
  if (i < NNODES * HID) out[i] = b[i & (HID - 1)];
}

// ---------------- segment softmax over incoming edges ----------------
__global__ void k_edge_max(const int* __restrict__ ei, const float* __restrict__ a_src,
                           const float* __restrict__ a_dst, float* maxb) {
  int i = blockIdx.x * blockDim.x + threadIdx.x;
  if (i >= ETOT * HEADS) return;
  int e = i >> 2, hh = i & 3, s, d;
  edge_sd(ei, e, s, d);
  float lg = a_src[s * HEADS + hh] + a_dst[d * HEADS + hh];
  lg = lg > 0.f ? lg : 0.2f * lg;                       // leaky_relu(0.2)
  atomic_max_f32(&maxb[d * HEADS + hh], lg);
}

__global__ void k_edge_expsum(const int* __restrict__ ei, const float* __restrict__ ew,
                              const float* __restrict__ a_src, const float* __restrict__ a_dst,
                              const float* __restrict__ maxb, float* sumb,
                              float* __restrict__ eedge, float* deg) {
  int i = blockIdx.x * blockDim.x + threadIdx.x;
  if (i >= ETOT * HEADS) return;
  int e = i >> 2, hh = i & 3, s, d;
  edge_sd(ei, e, s, d);
  float lg = a_src[s * HEADS + hh] + a_dst[d * HEADS + hh];
  lg = lg > 0.f ? lg : 0.2f * lg;
  float ex = __expf(lg - maxb[d * HEADS + hh]);
  eedge[i] = ex;
  atomicAdd(&sumb[d * HEADS + hh], ex);
  if (hh == 0) {                                        // fuse GCN degree accumulation
    float w = (e < NEDGES) ? ew[e] : 1.0f;
    atomicAdd(&deg[d], w);
  }
}

// ---------------- GAT message aggregation (scatter-add, L2 resident) ----------------
__global__ __launch_bounds__(256) void k_gat_agg(const int* __restrict__ ei,
    const float* __restrict__ h, const float* __restrict__ eedge,
    const float* __restrict__ sumb, float* out1)
{
  __shared__ float al[HEADS];
  int e = blockIdx.x, s, d;
  edge_sd(ei, e, s, d);
  if (threadIdx.x < HEADS)
    al[threadIdx.x] = eedge[e * HEADS + threadIdx.x] /
                      (sumb[d * HEADS + threadIdx.x] + 1e-16f);
  __syncthreads();
#pragma unroll
  for (int j = 0; j < 2; ++j) {
    int c = threadIdx.x + j * 256;
    atomicAdd(&out1[(size_t)d * HC + c], h[(size_t)s * HC + c] * al[c >> 7]);
  }
}

// ---------------- BatchNorm (training stats) + ReLU, emit bf16 for GEMM2 ----------------
__global__ __launch_bounds__(256) void k_bn_accum(const float* __restrict__ out1,
                                                  float* bnsum, float* bnsq) {
  int c = blockIdx.x * blockDim.x + threadIdx.x;        // column 0..511
  int chunk = (NNODES + gridDim.y - 1) / gridDim.y;
  int r0 = blockIdx.y * chunk;
  int r1 = r0 + chunk; if (r1 > NNODES) r1 = NNODES;
  float s = 0.f, q = 0.f;
  for (int r = r0; r < r1; ++r) {
    float v = out1[(size_t)r * HC + c];
    s += v; q += v * v;
  }
  atomicAdd(&bnsum[c], s);
  atomicAdd(&bnsq[c],  q);
}

__global__ void k_bn_final(const float* bnsum, const float* bnsq,
                           const float* __restrict__ gamma, const float* __restrict__ beta,
                           float* scale, float* shift) {
  int c = threadIdx.x;
  if (c < HC) {
    float mean = bnsum[c] * (1.0f / NNODES);
    float var  = bnsq[c]  * (1.0f / NNODES) - mean * mean;   // biased, like jnp.var
    float sc = gamma[c] * rsqrtf(var + 1e-5f);
    scale[c] = sc;
    shift[c] = beta[c] - mean * sc;
  }
}

__global__ void k_bn_apply(const float* __restrict__ out1, const float* __restrict__ scale,
                           const float* __restrict__ shift, unsigned short* __restrict__ xr) {
  int i = blockIdx.x * blockDim.x + threadIdx.x;
  if (i < NNODES * HC) {
    int c = i & (HC - 1);
    float v = out1[i] * scale[c] + shift[c];
    xr[i] = f2bf(v > 0.f ? v : 0.f);                    // ReLU
  }
}

// ---------------- GCN normalization + aggregation ----------------
__global__ void k_dinv(const float* __restrict__ deg, float* dinv) {
  int i = blockIdx.x * blockDim.x + threadIdx.x;
  if (i < NNODES) dinv[i] = deg[i] > 0.f ? rsqrtf(deg[i]) : 0.f;
}

__global__ __launch_bounds__(64) void k_gcn_agg(const int* __restrict__ ei,
    const float* __restrict__ ew, const float* __restrict__ h2,
    const float* __restrict__ dinv, float* out)
{
  __shared__ float snorm;
  int e = blockIdx.x, s, d;
  edge_sd(ei, e, s, d);
  if (threadIdx.x == 0) {
    float w = (e < NEDGES) ? ew[e] : 1.0f;
    snorm = dinv[s] * w * dinv[d];
  }
  __syncthreads();
  atomicAdd(&out[(size_t)d * HID + threadIdx.x],
            h2[(size_t)s * HID + threadIdx.x] * snorm);
}

// ---------------- launcher ----------------
extern "C" void kernel_launch(void* const* d_in, const int* in_sizes, int n_in,
                              void* d_out, int out_size, void* d_ws, size_t ws_size,
                              hipStream_t stream) {
  (void)in_sizes; (void)n_in; (void)out_size; (void)ws_size;
  const float* x     = (const float*)d_in[0];
  const int*   ei    = (const int*)  d_in[1];
  const float* ew    = (const float*)d_in[2];
  const float* Wg    = (const float*)d_in[3];
  const float* attS  = (const float*)d_in[4];
  const float* attD  = (const float*)d_in[5];
  const float* bgat  = (const float*)d_in[6];
  const float* gamma = (const float*)d_in[7];
  const float* beta  = (const float*)d_in[8];
  const float* Wc    = (const float*)d_in[9];
  const float* bgcn  = (const float*)d_in[10];
  float* out = (float*)d_out;

  // workspace layout with lifetime-based reuse
  char* ws = (char*)d_ws;
  size_t off = 0;
  auto take = [&](size_t b) -> char* {
    char* p = ws + off; off += (b + 255) & ~(size_t)255; return p;
  };
  char* regionA = take(76800000);   // phase1: xh(bf16 30000x1280) | phase2: out1 + eedge
  char* regionB = take(61440000);   // phase1: hfeat(f32 30000x512) | phase2: xr + h2
  unsigned short* xh    = (unsigned short*)regionA;
  float*          out1  = (float*)regionA;
  float*          eedge = (float*)(regionA + 61440000);   // (E+N)*4 f32 = 8.16MB
  float*          hfeat = (float*)regionB;
  unsigned short* xr    = (unsigned short*)regionB;
  float*          h2    = (float*)(regionB + 30720000);   // 30000*64 f32

  unsigned short* Wt   = (unsigned short*)take((size_t)HC  * F_INP * 2);
  unsigned short* Wt2  = (unsigned short*)take((size_t)HID * HC    * 2);
  float* a_src = (float*)take((size_t)NNODES * HEADS * 4);
  float* a_dst = (float*)take((size_t)NNODES * HEADS * 4);
  float* maxb  = (float*)take((size_t)NNODES * HEADS * 4);
  float* sumb  = (float*)take((size_t)NNODES * HEADS * 4);
  float* deg   = (float*)take((size_t)NNODES * 4);
  float* dinv  = (float*)take((size_t)NNODES * 4);
  float* bnsum = (float*)take(2048);
  float* bnsq  = (float*)take(2048);
  float* bnsc  = (float*)take(2048);
  float* bnsh  = (float*)take(2048);

  const int B = 256;
  auto cdiv = [](int a, int b) { return (a + b - 1) / b; };

  // 1) cast x -> bf16 ; transpose+cast weights for WMMA B operands
  k_cast_bf16<<<cdiv(NNODES * F_INP, B), B, 0, stream>>>(x, xh, NNODES * F_INP);
  k_transpose_cast<<<cdiv(HC  * F_INP, B), B, 0, stream>>>(Wg, Wt,  HC,  F_INP);
  k_transpose_cast<<<cdiv(HID * HC,    B), B, 0, stream>>>(Wc, Wt2, HID, HC);

  // 2) GEMM1: h = x @ W_gat (30000x1280 x 1280x512), WMMA bf16 + async LDS staging
  k_gemm_bf16_wmma<4, true><<<dim3(HC / 256, NNODES / 16), 128, 0, stream>>>(
      xh, Wt, hfeat, NNODES, HC, F_INP);

  // 3) attention logits per node
  k_att<<<NNODES, 512, 0, stream>>>(hfeat, attS, attD, a_src, a_dst);

  // 4) init accumulators (xh region now dead -> becomes out1/eedge)
  k_init_node<<<cdiv(NNODES * HEADS, B), B, 0, stream>>>(maxb, sumb, deg, bnsum, bnsq);
  k_init_out1<<<cdiv(NNODES * HC, B), B, 0, stream>>>(out1, bgat);

  // 5) segment softmax over incoming edges + degree
  k_edge_max<<<cdiv(ETOT * HEADS, B), B, 0, stream>>>(ei, a_src, a_dst, maxb);
  k_edge_expsum<<<cdiv(ETOT * HEADS, B), B, 0, stream>>>(ei, ew, a_src, a_dst,
                                                         maxb, sumb, eedge, deg);

  // 6) GAT message aggregation (scatter-add; out1 + hfeat fit in 192MB L2)
  k_gat_agg<<<ETOT, 256, 0, stream>>>(ei, hfeat, eedge, sumb, out1);

  // 7) BatchNorm stats + apply + ReLU -> bf16 xr (hfeat region now dead)
  k_bn_accum<<<dim3(HC / B, 64), B, 0, stream>>>(out1, bnsum, bnsq);
  k_bn_final<<<1, HC, 0, stream>>>(bnsum, bnsq, gamma, beta, bnsc, bnsh);
  k_bn_apply<<<cdiv(NNODES * HC, B), B, 0, stream>>>(out1, bnsc, bnsh, xr);

  // 8) GEMM2: h2 = xr @ W_gcn (30000x512 x 512x64) via WMMA, direct-register path
  k_gemm_bf16_wmma<1, false><<<dim3(1, NNODES / 16), 128, 0, stream>>>(
      xr, Wt2, h2, NNODES, HID, HC);

  // 9) GCN: symmetric norm + scatter aggregation
  k_dinv<<<cdiv(NNODES, B), B, 0, stream>>>(deg, dinv);
  k_init_out<<<cdiv(NNODES * HID, B), B, 0, stream>>>(out, bgcn);
  k_gcn_agg<<<ETOT, HID, 0, stream>>>(ei, ew, h2, dinv, out);
}
